// DecoderBlock_85916525789557
// MI455X (gfx1250) — matrix operand order repaired and checked by reading, not compiled, not run on hardware
//
#include <hip/hip_runtime.h>
#include <hip/hip_bf16.h>

typedef __attribute__((ext_vector_type(16))) _Float16 v16h;
typedef __attribute__((ext_vector_type(8)))  float    v8f;

#define NUM_HEADS 4
#define CDIM 64
#define HIDDEN 256
#define NWIN_TOTAL 32768   // 8 * (64*64) windows
#define NTILE_TOTAL 32768  // 524288 tokens / 16

__device__ __forceinline__ v8f wmma_f16(v16h a, v16h b, v8f c) {
  return __builtin_amdgcn_wmma_f32_16x16x32_f16(false, a, false, b, (short)0, c, false, false);
}

// A-fragment half-index -> K offset within a 32-wide chunk (ISA 16-bit A 16x32 table)
__device__ __forceinline__ int koffA(int p, int grp) {
  return 8 * grp + (p & 7) + ((p >> 3) << 4);
}
// B-fragment half-index -> K offset (same 16x32 table order as A)
__device__ __forceinline__ int koffB(int p, int grp) {
  return 8 * grp + (p & 7) + ((p >> 3) << 4);
}

// ---------------- Kernel 0: continuous relative position bias table -------------
// bias_tab[head*256 + i*16 + j] = 16*sigmoid( MLP(rel_coords(i,j))[head] )
__global__ void swin_cpb_kernel(const float* __restrict__ w1, const float* __restrict__ b1,
                                const float* __restrict__ w2, float* __restrict__ bias_tab) {
  int t = threadIdx.x;            // 256 threads: (i,j) token pairs
  int i = t >> 4, j = t & 15;
  int dh = (i >> 2) - (j >> 2);   // row delta in [-3,3]
  int dw = (i & 3) - (j & 3);
  float t0 = 8.0f * (float)dh / 3.0f;
  float t1 = 8.0f * (float)dw / 3.0f;
  t0 = copysignf(log2f(fabsf(t0) + 1.0f) / 3.0f, t0);
  t1 = copysignf(log2f(fabsf(t1) + 1.0f) / 3.0f, t1);
  float acc[NUM_HEADS] = {0.f, 0.f, 0.f, 0.f};
  for (int k = 0; k < 512; ++k) {
    float h = fmaxf(t0 * w1[2 * k] + t1 * w1[2 * k + 1] + b1[k], 0.0f);
#pragma unroll
    for (int hd = 0; hd < NUM_HEADS; ++hd) acc[hd] += h * w2[hd * 512 + k];
  }
#pragma unroll
  for (int hd = 0; hd < NUM_HEADS; ++hd)
    bias_tab[hd * 256 + t] = 16.0f / (1.0f + __expf(-acc[hd]));
}

// ---------------- Kernel 1: shifted-window attention + proj + LN residual ------
// One wave32 per 4x4 window (16 tokens x 64 ch). Writes x1 = x + LN(attn_out).
__global__ void swin_attn_kernel(const float* __restrict__ x,
                                 const float* __restrict__ qkv_w, const float* __restrict__ qkv_b,
                                 const float* __restrict__ proj_w, const float* __restrict__ proj_b,
                                 const float* __restrict__ logit_scale,
                                 const float* __restrict__ bias_tab,
                                 const float* __restrict__ n1g, const float* __restrict__ n1b,
                                 float* __restrict__ x1) {
  __shared__ _Float16 s_qkvw[192 * CDIM];     // 24 KB
  __shared__ _Float16 s_projw[CDIM * CDIM];   //  8 KB
  __shared__ float    s_qkvb[192];
  __shared__ float    s_projb[CDIM];
  __shared__ float    s_bias[NUM_HEADS * 256];
  __shared__ float    s_q[2][256], s_k[2][256], s_v[2][256];
  __shared__ _Float16 s_qn[2][256], s_kn[2][256];
  __shared__ float    s_attn[2][256];
  __shared__ _Float16 s_out[2][16 * CDIM];
  __shared__ float    s_y[2][16 * CDIM];

  const int tid = threadIdx.x;
  for (int i = tid; i < 192 * CDIM; i += blockDim.x) s_qkvw[i] = (_Float16)qkv_w[i];
  for (int i = tid; i < CDIM * CDIM; i += blockDim.x) s_projw[i] = (_Float16)proj_w[i];
  for (int i = tid; i < 192; i += blockDim.x)
    s_qkvb[i] = (i >= CDIM && i < 2 * CDIM) ? 0.0f : qkv_b[i];  // v2 zeroes k bias
  for (int i = tid; i < CDIM; i += blockDim.x) s_projb[i] = proj_b[i];
  for (int i = tid; i < NUM_HEADS * 256; i += blockDim.x) s_bias[i] = bias_tab[i];
  __syncthreads();

  const int wv = tid >> 5;
  const int lane = tid & 31;
  const int grp = lane >> 4;
  const int lm = lane & 15;

  float scale[NUM_HEADS];
#pragma unroll
  for (int h = 0; h < NUM_HEADS; ++h)
    scale[h] = __expf(fminf(logit_scale[h], 4.605170185988091f));  // log(100)

  const int gw = blockIdx.x * (blockDim.x >> 5) + wv;
  const int nwaves = gridDim.x * (blockDim.x >> 5);

  for (int wid = gw; wid < NWIN_TOTAL; wid += nwaves) {
    const int b = wid >> 12;
    const int wl = wid & 4095;
    const int wh = wl >> 6, ww = wl & 63;
    // lane lm <-> token row; shift fold: shifted (hs,ws) == original (hs+8, ws+8) mod 256
    const int hh = ((wh * 4 + (lm >> 2)) + 8) & 255;
    const int wc = ((ww * 4 + (lm & 3)) + 8) & 255;
    const int gbase = ((b * 256 + hh) * 256 + wc) * CDIM;

    // A fragments of X tile (16 x 64) in two K=32 chunks
    v16h a0, a1;
#pragma unroll
    for (int p = 0; p < 16; ++p) {
      const int k = koffA(p, grp);
      a0[p] = (_Float16)x[gbase + k];
      a1[p] = (_Float16)x[gbase + 32 + k];
    }

    for (int head = 0; head < NUM_HEADS; ++head) {
      // ---- q,k,v tiles for this head via WMMA ----
#pragma unroll
      for (int which = 0; which < 3; ++which) {
        const int t = which * 4 + head;  // output column tile of QKV (16 cols)
        const int n = t * 16 + lm;       // weight row (= output column)
        v16h b0, b1;
#pragma unroll
        for (int p = 0; p < 16; ++p) {
          const int k = koffB(p, grp);
          b0[p] = s_qkvw[n * CDIM + k];
          b1[p] = s_qkvw[n * CDIM + 32 + k];
        }
        v8f acc = {};
        acc = wmma_f16(a0, b0, acc);
        acc = wmma_f16(a1, b1, acc);
        const float cb = s_qkvb[t * 16 + lm];
        float* dst = (which == 0) ? s_q[wv] : (which == 1) ? s_k[wv] : s_v[wv];
#pragma unroll
        for (int v = 0; v < 8; ++v) dst[(v + 8 * grp) * 16 + lm] = acc[v] + cb;
      }
      __syncthreads();

      // ---- cosine-normalize rows: lanes 0-15 do q, 16-31 do k ----
      {
        const float* src = (grp == 0) ? s_q[wv] : s_k[wv];
        _Float16* dstn = (grp == 0) ? s_qn[wv] : s_kn[wv];
        float ss = 0.0f;
#pragma unroll
        for (int d = 0; d < 16; ++d) { float t = src[lm * 16 + d]; ss += t * t; }
        const float inv = 1.0f / fmaxf(sqrtf(ss), 1e-12f);
#pragma unroll
        for (int d = 0; d < 16; ++d) dstn[lm * 16 + d] = (_Float16)(src[lm * 16 + d] * inv);
      }
      __syncthreads();

      // ---- attn = qn @ kn^T (K=16 zero-padded to 32) ----
      v16h aq, bk;
#pragma unroll
      for (int p = 0; p < 16; ++p) {
        const int ka = koffA(p, grp);
        const int kb = koffB(p, grp);
        aq[p] = (ka < 16) ? s_qn[wv][lm * 16 + ka] : (_Float16)0.0f;
        bk[p] = (kb < 16) ? s_kn[wv][lm * 16 + kb] : (_Float16)0.0f;
      }
      v8f att = {};
      att = wmma_f16(aq, bk, att);

      // scale + CPB bias (shift-window mask is identically zero: 248 is 4-aligned)
#pragma unroll
      for (int v = 0; v < 8; ++v) {
        const int m = v + 8 * grp;
        att[v] = att[v] * scale[head] + s_bias[head * 256 + m * 16 + lm];
      }
      // row-wise softmax: each row lives across the 16 lanes of one half-wave
#pragma unroll
      for (int v = 0; v < 8; ++v) {
        float mx = att[v];
#pragma unroll
        for (int d = 1; d < 16; d <<= 1) mx = fmaxf(mx, __shfl_xor(mx, d, 16));
        const float e = __expf(att[v] - mx);
        float sm = e;
#pragma unroll
        for (int d = 1; d < 16; d <<= 1) sm += __shfl_xor(sm, d, 16);
        att[v] = e / sm;
      }
#pragma unroll
      for (int v = 0; v < 8; ++v) s_attn[wv][(v + 8 * grp) * 16 + lm] = att[v];
      __syncthreads();

      // ---- out_head = attn @ v (K=16 zero-padded) ----
      v16h aa, bv;
#pragma unroll
      for (int p = 0; p < 16; ++p) {
        const int ka = koffA(p, grp);
        const int kb = koffB(p, grp);
        aa[p] = (ka < 16) ? (_Float16)s_attn[wv][lm * 16 + ka] : (_Float16)0.0f;
        bv[p] = (kb < 16) ? (_Float16)s_v[wv][kb * 16 + lm] : (_Float16)0.0f;
      }
      v8f oh = {};
      oh = wmma_f16(aa, bv, oh);
#pragma unroll
      for (int v = 0; v < 8; ++v)
        s_out[wv][(v + 8 * grp) * CDIM + head * 16 + lm] = (_Float16)oh[v];
      __syncthreads();
    }

    // ---- projection: y = out @ proj_w^T + proj_b ----
    v16h ao0, ao1;
#pragma unroll
    for (int p = 0; p < 16; ++p) {
      const int k = koffA(p, grp);
      ao0[p] = s_out[wv][lm * CDIM + k];
      ao1[p] = s_out[wv][lm * CDIM + 32 + k];
    }
#pragma unroll
    for (int t = 0; t < 4; ++t) {
      const int n = t * 16 + lm;
      v16h b0, b1;
#pragma unroll
      for (int p = 0; p < 16; ++p) {
        const int k = koffB(p, grp);
        b0[p] = s_projw[n * CDIM + k];
        b1[p] = s_projw[n * CDIM + 32 + k];
      }
      v8f acc = {};
      acc = wmma_f16(ao0, b0, acc);
      acc = wmma_f16(ao1, b1, acc);
      const float pb = s_projb[t * 16 + lm];
#pragma unroll
      for (int v = 0; v < 8; ++v) s_y[wv][(v + 8 * grp) * CDIM + t * 16 + lm] = acc[v] + pb;
    }
    __syncthreads();

    // ---- x1 = x + LayerNorm(attn_out): lanes 0-15 each own one token ----
    if (grp == 0) {
      float mean = 0.0f;
#pragma unroll
      for (int c = 0; c < CDIM; ++c) mean += s_y[wv][lm * CDIM + c];
      mean *= (1.0f / CDIM);
      float var = 0.0f;
#pragma unroll
      for (int c = 0; c < CDIM; ++c) {
        const float d = s_y[wv][lm * CDIM + c] - mean;
        var += d * d;
      }
      const float inv = rsqrtf(var * (1.0f / CDIM) + 1e-5f);
#pragma unroll
      for (int c = 0; c < CDIM; ++c) {
        const float yn = (s_y[wv][lm * CDIM + c] - mean) * inv;
        x1[gbase + c] = x[gbase + c] + yn * n1g[c] + n1b[c];
      }
    }
    __syncthreads();
  }
}

// ---------------- Kernel 2: MLP (fc1+GELU+fc2) + LN residual -------------------
// One wave32 per 16-token tile. out = x1 + LN(gelu(x1 W1^T + b1) W2^T + b2)
__global__ void swin_mlp_kernel(const float* __restrict__ x1,
                                const float* __restrict__ fc1_w, const float* __restrict__ fc1_b,
                                const float* __restrict__ fc2_w, const float* __restrict__ fc2_b,
                                const float* __restrict__ n2g, const float* __restrict__ n2b,
                                float* __restrict__ out) {
  __shared__ _Float16 s_fc1w[HIDDEN * CDIM];  // 32 KB
  __shared__ float    s_fc1b[HIDDEN];
  __shared__ float    s_fc2b[CDIM];
  __shared__ _Float16 s_h[2][16 * HIDDEN];    // 16 KB
  __shared__ float    s_y[2][16 * CDIM];      //  8 KB

  const int tid = threadIdx.x;
  for (int i = tid; i < HIDDEN * CDIM; i += blockDim.x) s_fc1w[i] = (_Float16)fc1_w[i];
  for (int i = tid; i < HIDDEN; i += blockDim.x) s_fc1b[i] = fc1_b[i];
  for (int i = tid; i < CDIM; i += blockDim.x) s_fc2b[i] = fc2_b[i];
  __syncthreads();

  const int wv = tid >> 5;
  const int lane = tid & 31;
  const int grp = lane >> 4;
  const int lm = lane & 15;

  const int gw = blockIdx.x * (blockDim.x >> 5) + wv;
  const int nwaves = gridDim.x * (blockDim.x >> 5);

  for (int tile = gw; tile < NTILE_TOTAL; tile += nwaves) {
    const int base = (tile * 16 + lm) * CDIM;  // this lane's token row

    // A fragments of x1 tile (16 x 64)
    v16h a0, a1;
#pragma unroll
    for (int p = 0; p < 16; ++p) {
      const int k = koffA(p, grp);
      a0[p] = (_Float16)x1[base + k];
      a1[p] = (_Float16)x1[base + 32 + k];
    }

    // ---- h = gelu(x1 @ fc1_w^T + fc1_b), 16 column tiles ----
#pragma unroll
    for (int t = 0; t < 16; ++t) {
      const int n = t * 16 + lm;
      v16h b0, b1;
#pragma unroll
      for (int p = 0; p < 16; ++p) {
        const int k = koffB(p, grp);
        b0[p] = s_fc1w[n * CDIM + k];
        b1[p] = s_fc1w[n * CDIM + 32 + k];
      }
      v8f acc = {};
      acc = wmma_f16(a0, b0, acc);
      acc = wmma_f16(a1, b1, acc);
      const float cb = s_fc1b[t * 16 + lm];
#pragma unroll
      for (int v = 0; v < 8; ++v) {
        float g = acc[v] + cb;
        g = 0.5f * g * (1.0f + erff(g * 0.70710678118654752f));  // exact GELU
        s_h[wv][(v + 8 * grp) * HIDDEN + t * 16 + lm] = (_Float16)g;
      }
    }
    __syncthreads();

    // ---- y = h @ fc2_w^T + fc2_b (K=256 in 8 chunks), 4 column tiles ----
#pragma unroll
    for (int t = 0; t < 4; ++t) {
      const int n = t * 16 + lm;
      v8f acc = {};
#pragma unroll
      for (int kc = 0; kc < 8; ++kc) {
        v16h ah, bw;
#pragma unroll
        for (int p = 0; p < 16; ++p) {
          ah[p] = s_h[wv][lm * HIDDEN + 32 * kc + koffA(p, grp)];
          bw[p] = (_Float16)fc2_w[n * HIDDEN + 32 * kc + koffB(p, grp)];
        }
        acc = wmma_f16(ah, bw, acc);
      }
      const float cb = s_fc2b[t * 16 + lm];
#pragma unroll
      for (int v = 0; v < 8; ++v) s_y[wv][(v + 8 * grp) * CDIM + t * 16 + lm] = acc[v] + cb;
    }
    __syncthreads();

    // ---- out = x1 + LayerNorm(y): lanes 0-15 each own one token ----
    if (grp == 0) {
      float mean = 0.0f;
#pragma unroll
      for (int c = 0; c < CDIM; ++c) mean += s_y[wv][lm * CDIM + c];
      mean *= (1.0f / CDIM);
      float var = 0.0f;
#pragma unroll
      for (int c = 0; c < CDIM; ++c) {
        const float d = s_y[wv][lm * CDIM + c] - mean;
        var += d * d;
      }
      const float inv = rsqrtf(var * (1.0f / CDIM) + 1e-5f);
#pragma unroll
      for (int c = 0; c < CDIM; ++c) {
        const float yn = (s_y[wv][lm * CDIM + c] - mean) * inv;
        out[base + c] = x1[base + c] + yn * n2g[c] + n2b[c];
      }
    }
    __syncthreads();
  }
}

extern "C" void kernel_launch(void* const* d_in, const int* in_sizes, int n_in,
                              void* d_out, int out_size, void* d_ws, size_t ws_size,
                              hipStream_t stream) {
  const float* x           = (const float*)d_in[0];
  const float* qkv_w       = (const float*)d_in[1];
  const float* qkv_b       = (const float*)d_in[2];
  const float* proj_w      = (const float*)d_in[3];
  const float* proj_b      = (const float*)d_in[4];
  const float* logit_scale = (const float*)d_in[5];
  const float* cpb_w1      = (const float*)d_in[6];
  const float* cpb_b1      = (const float*)d_in[7];
  const float* cpb_w2      = (const float*)d_in[8];
  const float* norm1_g     = (const float*)d_in[9];
  const float* norm1_b     = (const float*)d_in[10];
  const float* norm2_g     = (const float*)d_in[11];
  const float* norm2_b     = (const float*)d_in[12];
  const float* fc1_w       = (const float*)d_in[13];
  const float* fc1_b       = (const float*)d_in[14];
  const float* fc2_w       = (const float*)d_in[15];
  const float* fc2_b       = (const float*)d_in[16];
  float* out = (float*)d_out;

  float* ws = (float*)d_ws;
  float* bias_tab = ws;            // 4 KB: [4][16][16] CPB bias
  float* x1 = ws + 1024;           // 134 MB: x + LN(attn) intermediate

  swin_cpb_kernel<<<1, 256, 0, stream>>>(cpb_w1, cpb_b1, cpb_w2, bias_tab);
  swin_attn_kernel<<<2048, 64, 0, stream>>>(x, qkv_w, qkv_b, proj_w, proj_b, logit_scale,
                                            bias_tab, norm1_g, norm1_b, x1);
  swin_mlp_kernel<<<2048, 64, 0, stream>>>(x1, fc1_w, fc1_b, fc2_w, fc2_b,
                                           norm2_g, norm2_b, out);
}
